// Net_30288109371815
// MI455X (gfx1250) — compile-verified, matrix operand order checked by reading, not run on hardware
//
#include <hip/hip_runtime.h>
#include <hip/hip_bf16.h>

typedef __attribute__((ext_vector_type(16))) __bf16 v16bf;
typedef __attribute__((ext_vector_type(8)))  float  v8f;

#define D_IN  256
#define H_DIM 32
#define C_DIM 2

// ---------------------------------------------------------------- utilities
__global__ void fill_kernel(float* __restrict__ p, long long n, float v) {
  long long i = (long long)blockIdx.x * blockDim.x + threadIdx.x;
  if (i < n) p[i] = v;
}

// deg[t] += 1 per edge (deg pre-initialized to 1.0 for the self-loop)
__global__ void degree_kernel(const int* __restrict__ ei, float* __restrict__ deg, int E) {
  int e = blockIdx.x * blockDim.x + threadIdx.x;
  if (e < E) atomicAdd(&deg[ei[E + e]], 1.0f);
}

__global__ void rsqrt_kernel(float* __restrict__ deg, int N) {
  int i = blockIdx.x * blockDim.x + threadIdx.x;
  if (i < N) {
    float d = deg[i];
    deg[i] = d > 0.0f ? rsqrtf(d) : 0.0f;
  }
}

// ---------------------------------------------------------------- GEMM1 (WMMA)
// t1 = x @ W1   [N x 256] @ [256 x 32] -> [N x 32]
// One wave per 16-row tile; 8 waves (256 threads) per block = 128 rows/block.
// K=256 -> 8 wmma steps of K=32; N=32 -> two 16x16 C tiles per wave.
__global__ void gemm1_wmma_kernel(const float* __restrict__ x,
                                  const float* __restrict__ W1,
                                  float* __restrict__ t1, int N) {
  // B fragments of W1, pre-swizzled to WMMA layout: [kstep][ntile][lane][16 halves]
  __shared__ __align__(32) __bf16 Bfrag[8][2][32][16];   // 16 KB LDS

  const int tid  = threadIdx.x;
  const int lane = tid & 31;
  const int wave = tid >> 5;

  // Stage W1 (K-major [256][32]) into WMMA B-fragment layout (mirrors A layout,
  // ISA 05_wmma "16-bit A-Matrix 16x32": K = 16*(jj/4) + 8*(lane/16) + 2*(jj&3) (+pair))
  for (int idx = tid; idx < 8 * 2 * 32 * 16; idx += blockDim.x) {
    int j     = idx & 15;         // half index within lane fragment
    int ln    = (idx >> 4) & 31;  // lane
    int ntile = (idx >> 9) & 1;   // which 16-col tile of the 32 outputs
    int ks    = idx >> 10;        // k-step 0..7
    int jj = j >> 1, p = j & 1;
    int hi = ln >> 4;
    int K  = ks * 32 + 16 * (jj >> 2) + 8 * hi + 2 * (jj & 3) + p;
    int n  = ntile * 16 + (ln & 15);
    Bfrag[ks][ntile][ln][j] = (__bf16)W1[K * H_DIM + n];
  }
  __syncthreads();

  long long rowbase = ((long long)blockIdx.x * 8 + wave) * 16;
  if (rowbase >= N) return;

  const int m  = lane & 15;   // C col within tile / A row within tile
  const int hi = lane >> 4;
  const float* xrow = x + (rowbase + m) * D_IN;

  v8f acc0 = {}; v8f acc1 = {};
  #pragma unroll
  for (int ks = 0; ks < 8; ++ks) {
    v16bf a;
    #pragma unroll
    for (int jj = 0; jj < 8; ++jj) {
      int K0 = ks * 32 + 16 * (jj >> 2) + 8 * hi + 2 * (jj & 3);
      float2 f = *(const float2*)(xrow + K0);       // global_load_b64
      a[2 * jj]     = (__bf16)f.x;
      a[2 * jj + 1] = (__bf16)f.y;
    }
    v16bf b0 = *(const v16bf*)&Bfrag[ks][0][lane][0]; // 2x ds_load_b128
    v16bf b1 = *(const v16bf*)&Bfrag[ks][1][lane][0];
    acc0 = __builtin_amdgcn_wmma_f32_16x16x32_bf16(false, a, false, b0,
                                                   (short)0, acc0, false, false);
    acc1 = __builtin_amdgcn_wmma_f32_16x16x32_bf16(false, a, false, b1,
                                                   (short)0, acc1, false, false);
  }

  // C/D layout: VGPR r, lane -> row M = r + 8*(lane/16), col = lane%16+tile*16
  #pragma unroll
  for (int r = 0; r < 8; ++r) {
    long long row = rowbase + r + 8 * hi;
    t1[row * H_DIM + m]      = acc0[r];
    t1[row * H_DIM + 16 + m] = acc1[r];
  }
}

// ---------------------------------------------------------------- scatter L1
// agg1[t] += t1[s] * dinv[s]*dinv[t].  8 lanes per edge, float4 each.
// t1/agg1 are 12.8MB each -> L2-resident on MI455X (192MB L2).
__global__ void scatter1_kernel(const int* __restrict__ ei,
                                const float* __restrict__ t1,
                                const float* __restrict__ dinv,
                                float* __restrict__ agg1, int E) {
  long long gid = (long long)blockIdx.x * blockDim.x + threadIdx.x;
  long long e   = gid >> 3;
  int part      = (int)(gid & 7);
  if (e >= E) return;
  long long s = ei[e];
  long long t = ei[(long long)E + e];
  float w = dinv[s] * dinv[t];
  float4 v = *(const float4*)(t1 + s * H_DIM + part * 4);
  float* dst = agg1 + t * H_DIM + part * 4;
  atomicAdd(dst + 0, v.x * w);
  atomicAdd(dst + 1, v.y * w);
  atomicAdd(dst + 2, v.z * w);
  atomicAdd(dst + 3, v.w * w);
}

// h1 = relu(agg1 + dinv^2 * t1 + b1), written in place over agg1
__global__ void bias_relu_kernel(float* __restrict__ agg1,
                                 const float* __restrict__ t1,
                                 const float* __restrict__ dinv,
                                 const float* __restrict__ b1, int N) {
  long long i = (long long)blockIdx.x * blockDim.x + threadIdx.x;
  if (i >= (long long)N * H_DIM) return;
  long long node = i >> 5;
  int n = (int)(i & 31);
  float di = dinv[node];
  float v = agg1[i] + di * di * t1[i] + b1[n];
  agg1[i] = v > 0.0f ? v : 0.0f;
}

// ---------------------------------------------------------------- GEMM2 (tiny)
// h2 = h1 @ W2   [N x 32] @ [32 x 2]
__global__ void gemm2_kernel(const float* __restrict__ h1,
                             const float* __restrict__ W2,
                             float* __restrict__ h2, int N) {
  int i = blockIdx.x * blockDim.x + threadIdx.x;
  if (i >= N) return;
  const float* hr = h1 + (long long)i * H_DIM;
  float z0 = 0.0f, z1 = 0.0f;
  #pragma unroll
  for (int k = 0; k < H_DIM; ++k) {
    float h = hr[k];
    z0 += h * W2[k * C_DIM + 0];
    z1 += h * W2[k * C_DIM + 1];
  }
  h2[(long long)i * C_DIM + 0] = z0;
  h2[(long long)i * C_DIM + 1] = z1;
}

// ---------------------------------------------------------------- scatter L2
__global__ void scatter2_kernel(const int* __restrict__ ei,
                                const float* __restrict__ h2,
                                const float* __restrict__ dinv,
                                float* __restrict__ agg2, int E) {
  long long e = (long long)blockIdx.x * blockDim.x + threadIdx.x;
  if (e >= E) return;
  long long s = ei[e];
  long long t = ei[(long long)E + e];
  float w = dinv[s] * dinv[t];
  atomicAdd(&agg2[t * C_DIM + 0], h2[s * C_DIM + 0] * w);
  atomicAdd(&agg2[t * C_DIM + 1], h2[s * C_DIM + 1] * w);
}

// logits = agg2 + dinv^2*h2 + b2 ; out = log_softmax(logits)
__global__ void logsoftmax_kernel(const float* __restrict__ agg2,
                                  const float* __restrict__ h2,
                                  const float* __restrict__ dinv,
                                  const float* __restrict__ b2,
                                  float* __restrict__ out, int N) {
  int i = blockIdx.x * blockDim.x + threadIdx.x;
  if (i >= N) return;
  float di = dinv[i], sl = di * di;
  float z0 = agg2[(long long)i * 2 + 0] + sl * h2[(long long)i * 2 + 0] + b2[0];
  float z1 = agg2[(long long)i * 2 + 1] + sl * h2[(long long)i * 2 + 1] + b2[1];
  float mx  = fmaxf(z0, z1);
  float lse = mx + logf(expf(z0 - mx) + expf(z1 - mx));
  out[(long long)i * 2 + 0] = z0 - lse;
  out[(long long)i * 2 + 1] = z1 - lse;
}

// ---------------------------------------------------------------- launcher
extern "C" void kernel_launch(void* const* d_in, const int* in_sizes, int n_in,
                              void* d_out, int out_size, void* d_ws, size_t ws_size,
                              hipStream_t stream) {
  const float* x  = (const float*)d_in[0];
  const int*   ei = (const int*)d_in[1];   // edge_index [2][E]
  const float* W1 = (const float*)d_in[2];
  const float* b1 = (const float*)d_in[3];
  const float* W2 = (const float*)d_in[4];
  const float* b2 = (const float*)d_in[5];
  float* out = (float*)d_out;

  const int N = in_sizes[0] / D_IN;      // 100000
  const int E = in_sizes[1] / 2;         // 3200000

  // workspace layout (floats): dinv[N] | t1[N*32] | agg1[N*32] | h2[N*2] | agg2[N*2]
  float* ws   = (float*)d_ws;
  float* dinv = ws;
  float* t1   = dinv + N;
  float* agg1 = t1 + (long long)N * H_DIM;
  float* h2   = agg1 + (long long)N * H_DIM;
  float* agg2 = h2 + (long long)N * C_DIM;

  const int B = 256;
  // init: deg starts at 1.0 (self-loop); aggregation buffers at 0
  fill_kernel<<<(N + B - 1) / B, B, 0, stream>>>(dinv, N, 1.0f);
  fill_kernel<<<((long long)N * H_DIM + B - 1) / B, B, 0, stream>>>(agg1, (long long)N * H_DIM, 0.0f);
  fill_kernel<<<((long long)N * C_DIM + B - 1) / B, B, 0, stream>>>(agg2, (long long)N * C_DIM, 0.0f);

  degree_kernel<<<(E + B - 1) / B, B, 0, stream>>>(ei, dinv, E);
  rsqrt_kernel<<<(N + B - 1) / B, B, 0, stream>>>(dinv, N);

  // Layer 1: WMMA transform, L2-resident scatter, fused self-loop+bias+relu
  gemm1_wmma_kernel<<<(N + 127) / 128, B, 0, stream>>>(x, W1, t1, N);
  scatter1_kernel<<<((long long)E * 8 + B - 1) / B, B, 0, stream>>>(ei, t1, dinv, agg1, E);
  bias_relu_kernel<<<((long long)N * H_DIM + B - 1) / B, B, 0, stream>>>(agg1, t1, dinv, b1, N);

  // Layer 2: tiny transform, scatter, log-softmax
  gemm2_kernel<<<(N + B - 1) / B, B, 0, stream>>>(agg1, W2, h2, N);
  scatter2_kernel<<<(E + B - 1) / B, B, 0, stream>>>(ei, h2, dinv, agg2, E);
  logsoftmax_kernel<<<(N + B - 1) / B, B, 0, stream>>>(agg2, h2, dinv, b2, out, N);
}
